// HDSGNN_46291157516821
// MI455X (gfx1250) — matrix operand order repaired and checked by reading, not compiled, not run on hardware
//
#include <hip/hip_runtime.h>
#include <math.h>

#define Nn 50000
#define Ff 128
#define Hh 256
#define Cc 40
#define CcPad 48
#define Ee 800000
#define Jj 2

typedef __attribute__((ext_vector_type(16))) __bf16        v16bf;
typedef __attribute__((ext_vector_type(8)))  float         v8f;
typedef __attribute__((ext_vector_type(4)))  unsigned int  v4u;
typedef __attribute__((ext_vector_type(8)))  unsigned int  v8u;

// ---------------- degree / norm precompute ----------------
__global__ void deg_init_kernel(float* deg, int n) {
    int i = blockIdx.x * blockDim.x + threadIdx.x;
    if (i < n) deg[i] = 1.0f;   // self-loop
}

__global__ void deg_edge_kernel(const int* __restrict__ dst, float* deg, int e) {
    int i = blockIdx.x * blockDim.x + threadIdx.x;
    if (i < e) atomicAdd(&deg[dst[i]], 1.0f);
}

__global__ void deg_rsqrt_kernel(float* deg, int n) {
    int i = blockIdx.x * blockDim.x + threadIdx.x;
    if (i < n) deg[i] = rsqrtf(deg[i]);
}

// ---------------- f32 -> bf16 conversion ----------------
__global__ void f32_to_bf16_kernel(const float* __restrict__ in,
                                   __bf16* __restrict__ out, long n) {
    long i = (long)blockIdx.x * blockDim.x + threadIdx.x;
    if (i < n) out[i] = (__bf16)in[i];
}

// W (KxNcols f32, row-major) -> Wt (NcolsPad x K bf16), zero-padded columns
__global__ void transpose_w_bf16_kernel(const float* __restrict__ W,
                                        __bf16* __restrict__ Wt,
                                        int K, int Ncols, int NcolsPad) {
    long i = (long)blockIdx.x * blockDim.x + threadIdx.x;
    long tot = (long)NcolsPad * K;
    if (i >= tot) return;
    int col = (int)(i / K);
    int k   = (int)(i - (long)col * K);
    float v = (col < Ncols) ? W[(size_t)k * Ncols + col] : 0.0f;
    Wt[i] = (__bf16)v;
}

// ---------------- bf16 WMMA GEMM ----------------
// A: M x K bf16 row-major.  Bt: NcolsPad x K bf16 (transposed weights).
// One wave computes a 16 x (16*T) strip: one A fragment reused for T WMMAs.
// Fragments are straight b128 loads (layouts per ISA 7.12.2):
//   A lane: K runs {8*half..+7} and {16+8*half..+7}  -> two 16B loads
//   B lane: col = col0+t*16+lidx, K runs {8*half..+7} and {16+8*half..+7} -> two 16B loads (Bt contiguous)
template<int T>
__global__ void wmma_gemm_bf16t(const __bf16* __restrict__ A,
                                const __bf16* __restrict__ Bt,
                                int K,
                                float* __restrict__ outF, int ldf,
                                __bf16* __restrict__ outB, int ldb,
                                const float* __restrict__ bias,
                                int doRelu, int Ncols) {
    const int row0 = blockIdx.x * 16;
    const int col0 = blockIdx.y * (16 * T);
    const int lane = threadIdx.x & 31;
    const int half = lane >> 4;
    const int lidx = lane & 15;

    v8f acc[T];
#pragma unroll
    for (int t = 0; t < T; ++t) acc[t] = (v8f){};

    const __bf16* Arow = A + (size_t)(row0 + lidx) * K;
    for (int k0 = 0; k0 < K; k0 += 32) {
        const v4u* pa = (const v4u*)(Arow + k0 + 8 * half);
        v4u a0 = pa[0];   // bf16 k = k0+8*half   .. +7
        v4u a1 = pa[2];   // bf16 k = k0+16+8*half .. +7
        v8u ar;
#pragma unroll
        for (int q = 0; q < 4; ++q) { ar[q] = a0[q]; ar[4 + q] = a1[q]; }
        v16bf af = __builtin_bit_cast(v16bf, ar);

#pragma unroll
        for (int t = 0; t < T; ++t) {
            int col = col0 + t * 16 + lidx;
            const v4u* pb = (const v4u*)(Bt + (size_t)col * K + k0 + 8 * half);
            v4u b0 = pb[0];
            v4u b1 = pb[2];
            v8u br;
#pragma unroll
            for (int q = 0; q < 4; ++q) { br[q] = b0[q]; br[4 + q] = b1[q]; }
            v16bf bf = __builtin_bit_cast(v16bf, br);
            acc[t] = __builtin_amdgcn_wmma_f32_16x16x32_bf16(
                /*neg_a=*/false, af, /*neg_b=*/false, bf,
                /*c_mod=*/(short)0, acc[t], /*reuse_a=*/false, /*reuse_b=*/false);
        }
    }

#pragma unroll
    for (int t = 0; t < T; ++t) {
        int col = col0 + t * 16 + lidx;
        if (col < Ncols) {
            float bb = bias ? bias[col] : 0.0f;
#pragma unroll
            for (int r = 0; r < 8; ++r) {
                float v = acc[t][r] + bb;
                if (doRelu) v = fmaxf(v, 0.0f);
                int row = row0 + r + 8 * half;
                if (outF) outF[(size_t)row * ldf + col] = v;
                if (outB) outB[(size_t)row * ldb + col] = (__bf16)v;
            }
        }
    }
}

// ---------------- GCN aggregation ----------------
// agg[n][j] = bias[j] + m[n][j]*dinv[n]^2   (bias + self-loop term)
__global__ void agg_init_kernel(const float* __restrict__ m,
                                const float* __restrict__ bias,
                                const float* __restrict__ dinv,
                                float* __restrict__ agg,
                                int n, int Hc) {
    long i = (long)blockIdx.x * blockDim.x + threadIdx.x;
    long tot = (long)n * Hc;
    if (i >= tot) return;
    int node = (int)(i / Hc);
    int j    = (int)(i - (long)node * Hc);
    float di = dinv[node];
    agg[i] = bias[j] + m[i] * di * di;
}

// one block per edge: agg[dst] += m[src] * dinv[src]*dinv[dst]
__global__ void gcn_scatter_kernel(const float* __restrict__ m,
                                   const int* __restrict__ src,
                                   const int* __restrict__ dst,
                                   const float* __restrict__ dinv,
                                   float* __restrict__ agg,
                                   int Hc) {
    int e = blockIdx.x;
    int s = src[e], d = dst[e];
    float nrm = dinv[s] * dinv[d];
    const float* ms = m + (size_t)s * Hc;
    float* ad = agg + (size_t)d * Hc;
    for (int j = threadIdx.x; j < Hc; j += blockDim.x)
        atomicAdd(&ad[j], ms[j] * nrm);
}

// ---------------- lin-feature assembly (bf16 out) ----------------
// lf[n, j*F + f] = (ow ? order_weights[j % J] : 1) * feature[begin + j][n][f]
__global__ void build_lf_kernel(const float* __restrict__ feature,
                                const float* __restrict__ ordw,
                                __bf16* __restrict__ lf,
                                int count, int begin, int useOw) {
    long i = (long)blockIdx.x * blockDim.x + threadIdx.x;
    long tot = (long)Nn * count * Ff;
    if (i >= tot) return;
    int cf = count * Ff;
    int n = (int)(i / cf);
    int r = (int)(i - (long)n * cf);
    int j = r / Ff;
    int f = r - j * Ff;
    float s = useOw ? ordw[j % Jj] : 1.0f;
    lf[i] = (__bf16)(s * feature[((size_t)(begin + j) * Nn + n) * Ff + f]);
}

// comb[n, colOff + j] = bf16(relu(agg[n, j]))
__global__ void relu_copy_bf16_kernel(const float* __restrict__ agg,
                                      __bf16* __restrict__ out,
                                      int Hc, int ldo, int colOff) {
    long i = (long)blockIdx.x * blockDim.x + threadIdx.x;
    long tot = (long)Nn * Hc;
    if (i >= tot) return;
    int node = (int)(i / Hc);
    int j    = (int)(i - (long)node * Hc);
    out[(size_t)node * ldo + colOff + j] = (__bf16)fmaxf(agg[i], 0.0f);
}

// ---------------- log-softmax over C=40 classes ----------------
__global__ void log_softmax_kernel(const float* __restrict__ logits,
                                   float* __restrict__ out) {
    int n = blockIdx.x * blockDim.x + threadIdx.x;
    if (n >= Nn) return;
    const float* l = logits + (size_t)n * Cc;
    float mx = l[0];
#pragma unroll
    for (int j = 1; j < Cc; ++j) mx = fmaxf(mx, l[j]);
    float s = 0.0f;
#pragma unroll
    for (int j = 0; j < Cc; ++j) s += expf(l[j] - mx);
    float lse = mx + logf(s);
    float* o = out + (size_t)n * Cc;
#pragma unroll
    for (int j = 0; j < Cc; ++j) o[j] = l[j] - lse;
}

extern "C" void kernel_launch(void* const* d_in, const int* in_sizes, int n_in,
                              void* d_out, int out_size, void* d_ws, size_t ws_size,
                              hipStream_t stream) {
    (void)in_sizes; (void)n_in; (void)out_size; (void)ws_size;

    const float* x       = (const float*)d_in[0];
    const float* feature = (const float*)d_in[1];
    const int*   src     = (const int*)d_in[2];
    const int*   dst     = (const int*)d_in[3];
    const float* convW[3] = {(const float*)d_in[4], (const float*)d_in[6], (const float*)d_in[8]};
    const float* convB[3] = {(const float*)d_in[5], (const float*)d_in[7], (const float*)d_in[9]};
    const float* linW[3]  = {(const float*)d_in[10], (const float*)d_in[12], (const float*)d_in[14]};
    const float* linB[3]  = {(const float*)d_in[11], (const float*)d_in[13], (const float*)d_in[15]};
    const float* ordw     = (const float*)d_in[16];
    const float* clsW     = (const float*)d_in[17];
    const float* clsB     = (const float*)d_in[18];

    // workspace carve-up (float units; all offsets stay 16B-aligned)
    float* ws     = (float*)d_ws;
    float* dinv   = ws;             ws += Nn;
    float* mconv  = ws;             ws += (size_t)Nn * Hh;       // f32 GEMM out for scatter
    float* agg    = ws;             ws += (size_t)Nn * Hh;       // scatter accumulator
    __bf16* combbf = (__bf16*)ws;   ws += (size_t)Nn * Hh;       // Nn x 2H bf16
    __bf16* lfbuf  = (__bf16*)ws;   ws += (size_t)Nn * Hh;       // Nn x 4F bf16
    __bf16* xbf    = (__bf16*)ws;   ws += (size_t)Nn * (Ff / 2); // Nn x F bf16
    __bf16* wtb    = (__bf16*)ws;
    const int convWtK[3] = {Ff, 2 * Hh, 2 * Hh};
    const int linWtK[3]  = {Ff, 2 * Ff, 4 * Ff};
    __bf16* convWt[3]; __bf16* linWt[3]; __bf16* clsWt;
    {
        __bf16* p = wtb;
        for (int i = 0; i < 3; ++i) { convWt[i] = p; p += (size_t)Hh * convWtK[i]; }
        for (int i = 0; i < 3; ++i) { linWt[i]  = p; p += (size_t)Hh * linWtK[i];  }
        clsWt = p;  // CcPad x 2H
    }

    float* out    = (float*)d_out;                 // log_softmax(logits)
    float* logits = out + (size_t)Nn * Cc;         // raw logits (2nd tuple element)

    auto cdiv = [](long a, long b) { return (unsigned)((a + b - 1) / b); };

    // ---- degrees / rsqrt norms ----
    deg_init_kernel <<<cdiv(Nn, 256), 256, 0, stream>>>(dinv, Nn);
    deg_edge_kernel <<<cdiv(Ee, 256), 256, 0, stream>>>(dst, dinv, Ee);
    deg_rsqrt_kernel<<<cdiv(Nn, 256), 256, 0, stream>>>(dinv, Nn);

    // ---- one-time packing: x -> bf16, weights -> transposed bf16 ----
    f32_to_bf16_kernel<<<cdiv((long)Nn * Ff, 256), 256, 0, stream>>>(x, xbf, (long)Nn * Ff);
    for (int i = 0; i < 3; ++i) {
        transpose_w_bf16_kernel<<<cdiv((long)Hh * convWtK[i], 256), 256, 0, stream>>>(
            convW[i], convWt[i], convWtK[i], Hh, Hh);
        transpose_w_bf16_kernel<<<cdiv((long)Hh * linWtK[i], 256), 256, 0, stream>>>(
            linW[i], linWt[i], linWtK[i], Hh, Hh);
    }
    transpose_w_bf16_kernel<<<cdiv((long)CcPad * 2 * Hh, 256), 256, 0, stream>>>(
        clsW, clsWt, 2 * Hh, Cc, CcPad);

    const __bf16* combIn = xbf;
    int Kin   = Ff;
    int begin = 0;
    for (int i = 0; i < 3; ++i) {
        int count = 1 << i;  // J^i, J=2
        dim3 gc(Nn / 16, Hh / 64);  // T=4 -> 64 cols per wave

        // conv path: mconv = combIn @ convW[i]  (f32 out for scatter)
        wmma_gemm_bf16t<4><<<gc, 32, 0, stream>>>(combIn, convWt[i], Kin,
                                                  mconv, Hh, ((__bf16*)nullptr), 0,
                                                  nullptr, 0, Hh);
        agg_init_kernel<<<cdiv((long)Nn * Hh, 256), 256, 0, stream>>>(
            mconv, convB[i], dinv, agg, Nn, Hh);
        gcn_scatter_kernel<<<Ee, 256, 0, stream>>>(mconv, src, dst, dinv, agg, Hh);

        // lin path: lf @ linW[i], fused bias+ReLU, bf16 out into combbf[:, 0:H]
        build_lf_kernel<<<cdiv((long)Nn * count * Ff, 256), 256, 0, stream>>>(
            feature, ordw, lfbuf, count, begin, i > 0 ? 1 : 0);
        wmma_gemm_bf16t<4><<<gc, 32, 0, stream>>>(lfbuf, linWt[i], count * Ff,
                                                  (float*)nullptr, 0, combbf, 2 * Hh,
                                                  linB[i], 1, Hh);
        // convx = relu(agg) into combbf[:, H:2H]
        relu_copy_bf16_kernel<<<cdiv((long)Nn * Hh, 256), 256, 0, stream>>>(
            agg, combbf, Hh, 2 * Hh, Hh);

        combIn = combbf;
        Kin    = 2 * Hh;
        begin += count;
    }

    // ---- classifier gcn ----
    dim3 gcls(Nn / 16, 1);  // T=3 -> 48 padded cols, store-guarded to 40
    wmma_gemm_bf16t<3><<<gcls, 32, 0, stream>>>(combbf, clsWt, 2 * Hh,
                                                mconv, Cc, ((__bf16*)nullptr), 0,
                                                nullptr, 0, Cc);
    agg_init_kernel<<<cdiv((long)Nn * Cc, 256), 256, 0, stream>>>(
        mconv, clsB, dinv, logits, Nn, Cc);
    gcn_scatter_kernel<<<Ee, 64, 0, stream>>>(mconv, src, dst, dinv, logits, Cc);

    log_softmax_kernel<<<cdiv(Nn, 128), 128, 0, stream>>>(logits, out);
}
